// DeepSetFeat_2233382994387
// MI455X (gfx1250) — matrix operand order repaired and checked by reading, not compiled, not run on hardware
//
#include <hip/hip_runtime.h>
#include <hip/hip_bf16.h>

typedef __attribute__((ext_vector_type(16))) _Float16 v16h;
typedef __attribute__((ext_vector_type(8)))  _Float16 v8h;
typedef __attribute__((ext_vector_type(8)))  float    v8f;

#define NN    1000000
#define GG    250000
#define DIN   64
#define DOUT  128
#define SLOPE 0.2f

__device__ __forceinline__ float leaky(float x) { return x >= 0.f ? x : x * SLOPE; }

__device__ __forceinline__ v16h make_frag(v8h lo, v8h hi) {
  v16h a;
#pragma unroll
  for (int j = 0; j < 8; ++j) { a[j] = lo[j]; a[j + 8] = hi[j]; }
  return a;
}

// A-fragment (16x32 f16) for v_wmma_f32_16x16x32_f16 from a row-major f16
// matrix with leading dimension ldk. ISA layout: lanes 0-15 hold M=lane,
// K = k0+[0..7] (v0..3) and k0+[16..23] (v4..7); lanes 16-31 hold the
// K=+8 / K=+24 runs. Each run is 8 contiguous halves = one 16-byte load.
__device__ __forceinline__ v16h load_a_frag(const _Float16* __restrict__ base,
                                            int ldk, int k0, int lane) {
  int m = lane & 15, h = lane >> 4;
  const _Float16* p = base + (size_t)m * ldk + k0 + h * 8;
  v8h lo = *(const v8h*)p;
  v8h hi = *(const v8h*)(p + 16);
  return make_frag(lo, hi);
}

// GEMM of a 16 x (32*ksteps) row-major f16 A-tile against a pre-packed
// weight (B fragments stored as [kstep][nblock][lane][16 halves], staged
// in LDS). Produces 8 n-blocks of 16x16 f32 accumulators.
__device__ __forceinline__ void wave_gemm(const _Float16* __restrict__ A, int lda,
                                          const _Float16* __restrict__ Bp,
                                          int ksteps, v8f acc[8], int lane) {
  for (int t = 0; t < ksteps; ++t) {
    v16h a = load_a_frag(A, lda, t * 32, lane);
#pragma unroll
    for (int b = 0; b < 8; ++b) {
      v16h bf = *(const v16h*)(Bp + ((size_t)((t * 8 + b) * 32 + lane)) * 16);
      acc[b] = __builtin_amdgcn_wmma_f32_16x16x32_f16(
          false, a, false, bf, (short)0, acc[b], false, false);
    }
  }
}

__device__ __forceinline__ void zero_acc(v8f acc[8]) {
  v8f z = {0.f, 0.f, 0.f, 0.f, 0.f, 0.f, 0.f, 0.f};
#pragma unroll
  for (int b = 0; b < 8; ++b) acc[b] = z;
}

// Block-cooperative copy of a packed weight into LDS (16-byte chunks).
__device__ __forceinline__ void stage_to_lds(_Float16* __restrict__ dst,
                                             const _Float16* __restrict__ src,
                                             int nhalf) {
  int nvec = nhalf >> 3;
  for (int i = threadIdx.x; i < nvec; i += blockDim.x)
    *(v8h*)(dst + i * 8) = *(const v8h*)(src + i * 8);
}

// BN(scale,bias)+LeakyReLU epilogue, result written as f16 into a 16xDOUT
// LDS tile (row-major) so it can be re-fragmented as the next GEMM's A.
// C layout: vgpr r, lanes 0-15 -> (M=r, N=lane); lanes 16-31 -> (M=r+8).
__device__ __forceinline__ void epilogue_to_tile(const v8f acc[8],
                                                 const float* __restrict__ sc,
                                                 const float* __restrict__ bi,
                                                 _Float16* __restrict__ tile, int lane) {
  int nl = lane & 15, h = lane >> 4;
#pragma unroll
  for (int b = 0; b < 8; ++b) {
    int n = b * 16 + nl;
    float s = sc[n], bb = bi[n];
#pragma unroll
    for (int r = 0; r < 8; ++r) {
      int m = r + 8 * h;
      tile[m * DOUT + n] = (_Float16)leaky(acc[b][r] * s + bb);
    }
  }
}

// Copy a wave's 16xDOUT f16 tile (contiguous 2048 halves in LDS) to a
// contiguous global destination with coalesced 16-byte stores.
__device__ __forceinline__ void store_tile16(const _Float16* __restrict__ tl,
                                             _Float16* __restrict__ dst, int lane) {
#pragma unroll
  for (int i = 0; i < 8; ++i) {
    int off = lane * 64 + i * 8;
    *(v8h*)(dst + off) = *(const v8h*)(tl + off);
  }
}

// ---------------- prep kernels ----------------

__global__ void k_cvt_f16(const float* __restrict__ src, _Float16* __restrict__ dst, int n) {
  for (int i = blockIdx.x * blockDim.x + threadIdx.x; i < n; i += gridDim.x * blockDim.x)
    dst[i] = (_Float16)src[i];
}

__global__ void k_zero_u32(unsigned int* __restrict__ p, int n) {
  for (int i = blockIdx.x * blockDim.x + threadIdx.x; i < n; i += gridDim.x * blockDim.x)
    p[i] = 0u;
}

// Pack row-major f32 weight [K][N] into WMMA B-fragment order:
// frag(t,b)[lane][j] = w[t*32 + (j>>3)*16 + (lane>>4)*8 + (j&7)][b*16 + (lane&15)]
__global__ void k_pack_w(const float* __restrict__ w, _Float16* __restrict__ wp,
                         int K, int N) {
  int total = K * N;
  int nb = N >> 4;
  for (int idx = blockIdx.x * blockDim.x + threadIdx.x; idx < total;
       idx += gridDim.x * blockDim.x) {
    int j = idx & 15;
    int lane = (idx >> 4) & 31;
    int rest = idx >> 9;
    int b = rest % nb;
    int t = rest / nb;
    int k = t * 32 + (j >> 3) * 16 + (lane >> 4) * 8 + (j & 7);
    int n = b * 16 + (lane & 15);
    wp[idx] = (_Float16)w[k * N + n];
  }
}

__global__ void k_seg_expand(const int* __restrict__ csr, int* __restrict__ seg) {
  int g = blockIdx.x * blockDim.x + threadIdx.x;
  if (g < GG) {
    int e = csr[g + 1];
    for (int i = csr[g]; i < e; ++i) seg[i] = g;
  }
}

// ---------------- stage 1: per-element MLP + segment-max scatter ----------------

__global__ __launch_bounds__(256) void k_mlp1(
    const _Float16* __restrict__ xh,
    const _Float16* __restrict__ w11p, const float* __restrict__ s11, const float* __restrict__ b11,
    const _Float16* __restrict__ w12p, const float* __restrict__ s12, const float* __restrict__ b12,
    const int* __restrict__ seg,
    _Float16* __restrict__ x1h, unsigned int* __restrict__ pooled) {
  __shared__ __attribute__((aligned(32))) _Float16 tiles[8 * 16 * DOUT]; // 32 KB
  __shared__ __attribute__((aligned(32))) _Float16 sw1[DIN * DOUT];      // 16 KB
  __shared__ __attribute__((aligned(32))) _Float16 sw2[DOUT * DOUT];     // 32 KB

  stage_to_lds(sw1, w11p, DIN * DOUT);
  stage_to_lds(sw2, w12p, DOUT * DOUT);
  __syncthreads();

  int wave = threadIdx.x >> 5, lane = threadIdx.x & 31;
  int tile = blockIdx.x * 8 + wave;
  if (tile < NN / 16) {                  // wave-uniform: EXEC stays all-ones
    int r0 = tile * 16;
    _Float16* tl = tiles + wave * 16 * DOUT;

    v8f acc[8];
    zero_acc(acc);
    wave_gemm(xh + (size_t)r0 * DIN, DIN, sw1, DIN / 32, acc, lane);    // x @ w11
    epilogue_to_tile(acc, s11, b11, tl, lane);

    zero_acc(acc);
    wave_gemm(tl, DOUT, sw2, DOUT / 32, acc, lane);                     // h @ w12
    epilogue_to_tile(acc, s12, b12, tl, lane);
    store_tile16(tl, x1h + (size_t)r0 * DOUT, lane);                    // coalesced x1

    // segment-max scatter (order-preserving float -> uint keys; init 0)
    int nl = lane & 15, h = lane >> 4;
    int gidx[8];
#pragma unroll
    for (int r = 0; r < 8; ++r) gidx[r] = seg[r0 + r + 8 * h];
#pragma unroll
    for (int b = 0; b < 8; ++b) {
      int n = b * 16 + nl;
      float s = s12[n], bb = b12[n];
#pragma unroll
      for (int r = 0; r < 8; ++r) {
        float v = leaky(acc[b][r] * s + bb);
        unsigned ub = __float_as_uint(v);
        unsigned key = ((int)ub >= 0) ? (ub | 0x80000000u) : ~ub;
        atomicMax(&pooled[(size_t)gidx[r] * DOUT + n], key);
      }
    }
  }
}

// ---------------- stage 2: per-group (set) MLP ----------------

__global__ __launch_bounds__(256) void k_set(
    const unsigned int* __restrict__ pooled, const int* __restrict__ csr,
    const _Float16* __restrict__ w21p, const float* __restrict__ s21, const float* __restrict__ b21,
    const _Float16* __restrict__ w22p, const float* __restrict__ s22, const float* __restrict__ b22,
    _Float16* __restrict__ xseth) {
  __shared__ __attribute__((aligned(32))) _Float16 tiles[8 * 16 * DOUT]; // 32 KB
  __shared__ __attribute__((aligned(32))) _Float16 sw1[DOUT * DOUT];     // 32 KB
  __shared__ __attribute__((aligned(32))) _Float16 sw2[DOUT * DOUT];     // 32 KB

  stage_to_lds(sw1, w21p, DOUT * DOUT);
  stage_to_lds(sw2, w22p, DOUT * DOUT);
  __syncthreads();

  int wave = threadIdx.x >> 5, lane = threadIdx.x & 31;
  int tile = blockIdx.x * 8 + wave;
  if (tile < GG / 16) {
    int g0 = tile * 16;
    _Float16* tl = tiles + wave * 16 * DOUT;

    // cooperative decode of the pooled-max keys into an f16 A-tile
#pragma unroll 4
    for (int j = 0; j < 64; ++j) {
      int idx = lane * 64 + j;            // 16*128 = 2048 = 32 lanes * 64
      int m = idx >> 7, n = idx & 127;
      int g = g0 + m;
      int cnt = csr[g + 1] - csr[g];
      float f = 0.f;
      if (cnt > 0) {
        unsigned u = pooled[(size_t)g * DOUT + n];
        unsigned bits = (u & 0x80000000u) ? (u ^ 0x80000000u) : ~u;
        f = __uint_as_float(bits);
      }
      tl[idx] = (_Float16)f;
    }

    v8f acc[8];
    zero_acc(acc);
    wave_gemm(tl, DOUT, sw1, DOUT / 32, acc, lane);
    epilogue_to_tile(acc, s21, b21, tl, lane);

    zero_acc(acc);
    wave_gemm(tl, DOUT, sw2, DOUT / 32, acc, lane);
    epilogue_to_tile(acc, s22, b22, tl, lane);
    store_tile16(tl, xseth + (size_t)g0 * DOUT, lane);                  // coalesced
  }
}

// ---------------- stage 3: concat fusion + final MLP ----------------

__global__ __launch_bounds__(256) void k_final(
    const _Float16* __restrict__ x1h, const _Float16* __restrict__ xseth,
    const int* __restrict__ seg,
    const _Float16* __restrict__ w31p, const float* __restrict__ s31, const float* __restrict__ b31,
    const _Float16* __restrict__ w32p, const float* __restrict__ s32, const float* __restrict__ b32,
    float* __restrict__ out) {
  __shared__ __attribute__((aligned(32))) _Float16 tiles[8 * 16 * DOUT];  // 32 KB
  __shared__ __attribute__((aligned(32))) _Float16 sw1[2 * DOUT * DOUT];  // 64 KB
  __shared__ __attribute__((aligned(32))) _Float16 sw2[DOUT * DOUT];      // 32 KB

  stage_to_lds(sw1, w31p, 2 * DOUT * DOUT);
  stage_to_lds(sw2, w32p, DOUT * DOUT);
  __syncthreads();

  int wave = threadIdx.x >> 5, lane = threadIdx.x & 31;
  int tile = blockIdx.x * 8 + wave;
  if (tile < NN / 16) {
    int r0 = tile * 16;
    _Float16* tl = tiles + wave * 16 * DOUT;
    int nl = lane & 15, h = lane >> 4;

    v8f acc[8];
    zero_acc(acc);
    // concat([x1, xset[seg]]) @ w31  ==  x1@w31[0:128] + xset[seg]@w31[128:256]
    wave_gemm(x1h + (size_t)r0 * DOUT, DOUT, sw1, 4, acc, lane);

    const _Float16* arow = xseth + (size_t)seg[r0 + nl] * DOUT;  // per-lane row gather
    for (int t = 0; t < 4; ++t) {
      v8h lo = *(const v8h*)(arow + t * 32 + h * 8);
      v8h hi = *(const v8h*)(arow + t * 32 + h * 8 + 16);
      v16h a = make_frag(lo, hi);
#pragma unroll
      for (int b = 0; b < 8; ++b) {
        v16h bf = *(const v16h*)(sw1 + (size_t)(((t + 4) * 8 + b) * 32 + lane) * 16);
        acc[b] = __builtin_amdgcn_wmma_f32_16x16x32_f16(
            false, a, false, bf, (short)0, acc[b], false, false);
      }
    }
    epilogue_to_tile(acc, s31, b31, tl, lane);

    zero_acc(acc);
    wave_gemm(tl, DOUT, sw2, DOUT / 32, acc, lane);

#pragma unroll
    for (int b = 0; b < 8; ++b) {
      int n = b * 16 + nl;
      float s = s32[n], bb = b32[n];
#pragma unroll
      for (int r = 0; r < 8; ++r) {
        int m = r + 8 * h;
        out[(size_t)(r0 + m) * DOUT + n] = leaky(acc[b][r] * s + bb);
      }
    }
  }
}

// ---------------- host launcher ----------------

extern "C" void kernel_launch(void* const* d_in, const int* in_sizes, int n_in,
                              void* d_out, int out_size, void* d_ws, size_t ws_size,
                              hipStream_t stream) {
  const float* x    = (const float*)d_in[0];
  const int*   csr  = (const int*)d_in[1];
  const float* w11  = (const float*)d_in[2];
  const float* s11  = (const float*)d_in[3];
  const float* b11  = (const float*)d_in[4];
  const float* w12  = (const float*)d_in[5];
  const float* s12  = (const float*)d_in[6];
  const float* b12  = (const float*)d_in[7];
  const float* w21  = (const float*)d_in[8];
  const float* s21  = (const float*)d_in[9];
  const float* b21  = (const float*)d_in[10];
  const float* w22  = (const float*)d_in[11];
  const float* s22  = (const float*)d_in[12];
  const float* b22  = (const float*)d_in[13];
  const float* w31  = (const float*)d_in[14];
  const float* s31  = (const float*)d_in[15];
  const float* b31  = (const float*)d_in[16];
  const float* w32  = (const float*)d_in[17];
  const float* s32  = (const float*)d_in[18];
  const float* b32  = (const float*)d_in[19];
  float* out = (float*)d_out;

  // workspace carve-up (all offsets keep >=256B alignment)
  char* p = (char*)d_ws;
  _Float16* xh    = (_Float16*)p;  p += (size_t)NN * DIN  * 2;   // x in f16
  _Float16* x1h   = (_Float16*)p;  p += (size_t)NN * DOUT * 2;   // stage-1 out, f16
  _Float16* xseth = (_Float16*)p;  p += (size_t)GG * DOUT * 2;   // stage-2 out, f16
  unsigned* pool  = (unsigned*)p;  p += (size_t)GG * DOUT * 4;   // segment-max keys
  int*      seg   = (int*)p;       p += (size_t)NN * 4;          // element -> group
  _Float16* w11p  = (_Float16*)p;  p += (size_t)DIN  * DOUT * 2;
  _Float16* w12p  = (_Float16*)p;  p += (size_t)DOUT * DOUT * 2;
  _Float16* w21p  = (_Float16*)p;  p += (size_t)DOUT * DOUT * 2;
  _Float16* w22p  = (_Float16*)p;  p += (size_t)DOUT * DOUT * 2;
  _Float16* w31p  = (_Float16*)p;  p += (size_t)2 * DOUT * DOUT * 2;
  _Float16* w32p  = (_Float16*)p;  p += (size_t)DOUT * DOUT * 2;

  // prep
  k_cvt_f16<<<4096, 256, 0, stream>>>(x, xh, NN * DIN);
  k_zero_u32<<<4096, 256, 0, stream>>>(pool, GG * DOUT);
  k_seg_expand<<<(GG + 255) / 256, 256, 0, stream>>>(csr, seg);
  k_pack_w<<<32,  256, 0, stream>>>(w11, w11p, DIN,  DOUT);
  k_pack_w<<<64,  256, 0, stream>>>(w12, w12p, DOUT, DOUT);
  k_pack_w<<<64,  256, 0, stream>>>(w21, w21p, DOUT, DOUT);
  k_pack_w<<<64,  256, 0, stream>>>(w22, w22p, DOUT, DOUT);
  k_pack_w<<<128, 256, 0, stream>>>(w31, w31p, 2 * DOUT, DOUT);
  k_pack_w<<<64,  256, 0, stream>>>(w32, w32p, DOUT, DOUT);

  // pipeline (one wave per 16-row tile, 8 waves / 256-thread block)
  const int tiles_n = NN / 16;   // 62500
  const int tiles_g = GG / 16;   // 15625
  k_mlp1<<<(tiles_n + 7) / 8, 256, 0, stream>>>(xh, w11p, s11, b11, w12p, s12, b12,
                                                seg, x1h, pool);
  k_set<<<(tiles_g + 7) / 8, 256, 0, stream>>>(pool, csr, w21p, s21, b21,
                                               w22p, s22, b22, xseth);
  k_final<<<(tiles_n + 7) / 8, 256, 0, stream>>>(x1h, xseth, seg,
                                                 w31p, s31, b31, w32p, s32, b32, out);
}
